// OPQPQQuantizer_824633721394
// MI455X (gfx1250) — matrix operand order repaired and checked by reading, not compiled, not run on hardware
//
#include <hip/hip_runtime.h>
#include <hip/hip_bf16.h>
#include <stdint.h>

// ---------------- problem constants ----------------
constexpr int Bb = 4, Tt = 2048, Dd = 512, Mm = 8, Kk = 512, DSs = 64;
constexpr int Nn = Bb * Tt;                 // 8192 rows
// ---------------- workspace layout (bytes) ----------------
constexpr size_t ZROT_OFF = 0;                                   // f32 [N][D]  16 MB
constexpr size_t ZN_OFF   = ZROT_OFF + (size_t)Nn * Dd * 4;      // bf16 [N][D]  8 MB
constexpr size_t CNBF_OFF = ZN_OFF   + (size_t)Nn * Dd * 2;      // bf16 [M*K][DS]
constexpr size_t WBF_OFF  = CNBF_OFF + (size_t)Mm * Kk * DSs * 2;// bf16 [D][D]
constexpr size_t IDX_OFF  = WBF_OFF  + (size_t)Dd * Dd * 2;      // i32 [N][M]
constexpr size_t CNT_OFF  = IDX_OFF  + (size_t)Nn * Mm * 4;      // i32 [M][K]
constexpr size_t ACC_OFF  = CNT_OFF  + (size_t)Mm * Kk * 4;      // f32 [4]

// ---------------- types ----------------
typedef __attribute__((ext_vector_type(16))) __bf16 v16bf;
typedef __attribute__((ext_vector_type(8)))  float  v8f;
typedef __attribute__((ext_vector_type(4)))  unsigned int u32x4;
typedef __attribute__((ext_vector_type(8)))  int  i32x8;
typedef __attribute__((ext_vector_type(4)))  int  i32x4;

// native hardware convert (v_cvt_*bf16_f32) instead of bit-twiddled RNE emulation
static __device__ __forceinline__ __bf16 f32_to_bf16(float f) { return (__bf16)f; }
static __device__ __forceinline__ unsigned short f32_to_bf16_bits(float f) {
    union { __bf16 b; unsigned short u; } o; o.b = (__bf16)f; return o.u;
}

// ---------------- WMMA helpers (wave32, 16x16x32 bf16; k = 16*(lane/16) + s, not AMD's table order) ----------------
// A fragment: 16x32 (MxK). lanes 0-15: row=lane, K=0..15; lanes 16-31: K=16..31
static __device__ __forceinline__ v16bf load_a_frag(const __bf16* rowbase, int lda, int lane) {
    int r  = lane & 15;
    int kh = (lane >> 4) ? 16 : 0;
    const __bf16* p = rowbase + (size_t)r * lda + kh;
    union { v16bf v; uint4 q[2]; } f;
    f.q[0] = *(const uint4*)(p);
    f.q[1] = *(const uint4*)(p + 8);
    return f.v;
}
// B fragment: 32x16 (KxN), column n contiguous in K. lanes 0-15: n=lane, K=0..15; lanes 16-31: K=16..31
static __device__ __forceinline__ v16bf load_b_frag(const __bf16* colbase, int ldc, int lane) {
    int n  = lane & 15;
    int kh = (lane >> 4) ? 16 : 0;
    const __bf16* p = colbase + (size_t)n * ldc + kh;
    union { v16bf v; uint4 q[2]; } f;
    f.q[0] = *(const uint4*)(p);
    f.q[1] = *(const uint4*)(p + 8);
    return f.v;
}
// B fragment built from an f32 LDS tile Wt[32][64] (row-major), converting to bf16 per lane.
static __device__ __forceinline__ v16bf load_b_frag_f32(const float* Wt, int ncolbase, int lane) {
    int n  = ncolbase + (lane & 15);
    int kh = (lane >> 4) ? 16 : 0;
    union { v16bf v; unsigned short u[16]; } f;
#pragma unroll
    for (int kk = 0; kk < 16; ++kk)
        f.u[kk] = f32_to_bf16_bits(Wt[(size_t)(kh + kk) * 64 + n]);
    return f.v;
}
static __device__ __forceinline__ v8f wmma_bf16(v16bf a, v16bf b, v8f c) {
    return __builtin_amdgcn_wmma_f32_16x16x32_bf16(false, a, false, b, (short)0, c, false, false);
}

// ---------------- Tensor Data Mover: 2D tile -> LDS ----------------
static __device__ __forceinline__ void tdm_load_2d(uint32_t lds_off, const void* gaddr,
                                                   uint32_t tile_dim0, uint32_t tile_dim1,
                                                   uint64_t stride0_elems, int data_size_code) {
    uint64_t ga = (uint64_t)(uintptr_t)gaddr;
    u32x4 g0;
    g0[0] = 1u;                                         // count=1, user descriptor
    g0[1] = lds_off;                                    // LDS byte address
    g0[2] = (uint32_t)(ga & 0xFFFFFFFFu);               // global_addr[31:0]
    g0[3] = (uint32_t)((ga >> 32) & 0x01FFFFFFu) | 0x80000000u; // global_addr[56:32] | type=2
    const uint32_t dim0 = 1u << 20, dim1 = 1u << 20;    // generous (no OOB clipping needed)
    i32x8 g1;
    g1[0] = (int)(((uint32_t)data_size_code & 3u) << 16);               // mask=0, data_size
    g1[1] = (int)((dim0 & 0xFFFFu) << 16);                              // tensor_dim0[15:0]
    g1[2] = (int)(((dim0 >> 16) & 0xFFFFu) | ((dim1 & 0xFFFFu) << 16)); // dim0[31:16]|dim1[15:0]
    g1[3] = (int)(((dim1 >> 16) & 0xFFFFu) | ((tile_dim0 & 0xFFFFu) << 16));
    g1[4] = (int)(tile_dim1 & 0xFFFFu);                                 // tile_dim1, tile_dim2=0
    g1[5] = (int)(uint32_t)(stride0_elems & 0xFFFFFFFFu);               // tensor_dim0_stride lo
    g1[6] = (int)(uint32_t)((stride0_elems >> 32) & 0xFFFFu);           // stride hi
    g1[7] = 0;
    i32x4 z4 = {0, 0, 0, 0};
#if defined(__clang_major__) && (__clang_major__ >= 23)
    i32x8 z8 = {0, 0, 0, 0, 0, 0, 0, 0};
    __builtin_amdgcn_tensor_load_to_lds(g0, g1, z4, z4, z8, 0);
#else
    __builtin_amdgcn_tensor_load_to_lds(g0, g1, z4, z4, 0);
#endif
}

// ============== kernel 0: prep (normalized bf16 codebooks + bf16 W) ==============
__global__ void __launch_bounds__(256) opq_prep_kernel(const float* __restrict__ W,
                                                       const float* __restrict__ codebooks,
                                                       __bf16* __restrict__ cnbf,
                                                       __bf16* __restrict__ wbf) {
    int b = blockIdx.x, tid = threadIdx.x;
    if (b < 16) {                           // 4096 codewords, one per thread
        int c = b * 256 + tid;              // c = m*512 + k
        const float* src = codebooks + (size_t)c * DSs;
        float s = 0.f;
#pragma unroll 4
        for (int d = 0; d < DSs; ++d) { float v = src[d]; s += v * v; }
        float scale = 1.0f / fmaxf(sqrtf(s), 1e-12f);
        __bf16* dst = cnbf + (size_t)c * DSs;
#pragma unroll 4
        for (int d = 0; d < DSs; ++d) dst[d] = f32_to_bf16(src[d] * scale);
    } else {
        size_t p = (size_t)(b - 16) * 256 + tid;   // < 262144
        wbf[p] = f32_to_bf16(W[p]);
    }
}

// ============== kernel 1: z_rot = z @ W, then per-subspace normalize ==============
// grid (N/64, D/64)=(128,8), block 256 (8 waves). W k-panels staged to LDS via TDM.
__global__ void __launch_bounds__(256) opq_rotate_kernel(const float* __restrict__ z,
                                                         const float* __restrict__ W,
                                                         float* __restrict__ zrot,
                                                         __bf16* __restrict__ zn) {
    __shared__ float  Wt[32 * 64];       // TDM destination, f32 tile of W
    __shared__ __bf16 At[64 * 32];       // bf16 A tile
    __shared__ float  Zt[64 * 65];       // output tile (padded) for normalization
    __shared__ float  rnorm[64];

    int row0 = blockIdx.x * 64, col0 = blockIdx.y * 64;
    int tid = threadIdx.x, lane = tid & 31, wid = tid >> 5;
    int wr = wid >> 1, wc = wid & 1;     // wave -> 16-row strip, 32-col half

    v8f acc0 = {0, 0, 0, 0, 0, 0, 0, 0};
    v8f acc1 = acc0;

    for (int k0 = 0; k0 < Dd; k0 += 32) {
        if (wid == 0) {                  // wave 0 drives the TDM (EXEC ignored by tensor ops)
            tdm_load_2d((uint32_t)(uintptr_t)Wt, W + (size_t)k0 * Dd + col0,
                        /*tile_dim0=*/64, /*tile_dim1=*/32, /*stride0=*/Dd, /*dsize=*/2);
        }
        {   // all 256 threads stage A: 64 rows x 32 k, 8 f32 each -> bf16 LDS
            int ar = tid >> 2, ac = (tid & 3) * 8;
            const float* zp = z + (size_t)(row0 + ar) * Dd + k0 + ac;
            float4 f0 = *(const float4*)zp;
            float4 f1 = *(const float4*)(zp + 4);
            __bf16* ap = At + (size_t)ar * 32 + ac;
            ap[0] = f32_to_bf16(f0.x); ap[1] = f32_to_bf16(f0.y);
            ap[2] = f32_to_bf16(f0.z); ap[3] = f32_to_bf16(f0.w);
            ap[4] = f32_to_bf16(f1.x); ap[5] = f32_to_bf16(f1.y);
            ap[6] = f32_to_bf16(f1.z); ap[7] = f32_to_bf16(f1.w);
        }
        if (wid == 0) __builtin_amdgcn_s_wait_tensorcnt(0);
        __syncthreads();

        v16bf a  = load_a_frag(At + (size_t)wr * 16 * 32, 32, lane);
        v16bf b0 = load_b_frag_f32(Wt, wc * 32 + 0,  lane);
        v16bf b1 = load_b_frag_f32(Wt, wc * 32 + 16, lane);
        acc0 = wmma_bf16(a, b0, acc0);
        acc1 = wmma_bf16(a, b1, acc1);
        __syncthreads();                 // Wt/At consumed before next TDM overwrite
    }

    // scatter C fragments to LDS tile (C layout: slot r -> row r / r+8, col = lane&15)
    int rh = (lane >> 4) ? 8 : 0;
#pragma unroll
    for (int r = 0; r < 8; ++r) {
        int rr = wr * 16 + r + rh;
        Zt[(size_t)rr * 65 + wc * 32 + (lane & 15)]      = acc0[r];
        Zt[(size_t)rr * 65 + wc * 32 + 16 + (lane & 15)] = acc1[r];
    }
    __syncthreads();
    if (tid < 64) {                      // the 64-col tile is exactly one DS=64 subspace
        float s = 0.f;
#pragma unroll 4
        for (int c = 0; c < 64; ++c) { float v = Zt[(size_t)tid * 65 + c]; s += v * v; }
        rnorm[tid] = 1.0f / fmaxf(sqrtf(s), 1e-12f);
    }
    __syncthreads();
    {
        int rr = tid >> 2, cc = (tid & 3) * 16;
        float rn = rnorm[rr];
        size_t g = (size_t)(row0 + rr) * Dd + col0 + cc;
#pragma unroll 4
        for (int c = 0; c < 16; ++c) {
            float v = Zt[(size_t)rr * 65 + cc + c];
            zrot[g + c] = v;
            zn[g + c]   = f32_to_bf16(v * rn);
        }
    }
}

// ============== kernel 2: cos GEMM fused with gumbel + argmax (softmax eliminated) ==============
// one wave per (16-row block, subspace m). grid 1024, block 128.
__global__ void __launch_bounds__(128) opq_argmax_kernel(const __bf16* __restrict__ zn,
                                                         const __bf16* __restrict__ cnbf,
                                                         const float* __restrict__ gumbel,
                                                         int* __restrict__ idxbuf,
                                                         float* __restrict__ out_idx,
                                                         int* __restrict__ counts) {
    int tid = threadIdx.x, lane = tid & 31;
    int gw = blockIdx.x * 4 + (tid >> 5);         // 0..4095
    int r0 = (gw >> 3) * 16;
    int m  = gw & 7;

    const __bf16* znp = zn + (size_t)r0 * Dd + m * DSs;
    v16bf a0 = load_a_frag(znp,      Dd, lane);   // K 0..31 of subspace
    v16bf a1 = load_a_frag(znp + 32, Dd, lane);   // K 32..63
    const __bf16* cb = cnbf + (size_t)m * Kk * DSs;
    const float*  gp = gumbel + ((size_t)r0 * Mm + m) * Kk;

    float best[8]; int bk[8];
#pragma unroll
    for (int r = 0; r < 8; ++r) { best[r] = -1e30f; bk[r] = 0; }

    int rh = (lane >> 4) ? 8 : 0;
    for (int t = 0; t < 32; ++t) {                // 32 k-tiles of 16 codewords
        v16bf b0 = load_b_frag(cb + (size_t)t * 16 * DSs,      DSs, lane);
        v16bf b1 = load_b_frag(cb + (size_t)t * 16 * DSs + 32, DSs, lane);
        v8f c = {0, 0, 0, 0, 0, 0, 0, 0};
        c = wmma_bf16(a0, b0, c);
        c = wmma_bf16(a1, b1, c);
        int k = t * 16 + (lane & 15);
#pragma unroll
        for (int r = 0; r < 8; ++r) {
            float v = c[r] + gp[(size_t)(r + rh) * Mm * Kk + k];   // cos + gumbel
            if (v > best[r]) { best[r] = v; bk[r] = k; }
        }
    }
    // reduce within 16-lane halves (matches C-layout row split on wave32)
#pragma unroll
    for (int r = 0; r < 8; ++r) {
        float bv = best[r]; int bkk = bk[r];
        for (int off = 1; off < 16; off <<= 1) {
            float ov = __shfl_xor(bv, off, 16);
            int   ok = __shfl_xor(bkk, off, 16);
            if (ov > bv || (ov == bv && ok < bkk)) { bv = ov; bkk = ok; }
        }
        if ((lane & 15) == 0) {
            int row = r0 + r + rh;
            idxbuf[(size_t)row * Mm + m]  = bkk;
            out_idx[(size_t)row * Mm + m] = (float)bkk;
            atomicAdd(&counts[m * Kk + bkk], 1);
        }
    }
}

// ============== kernel 3: gather quantized + commitment loss + z_q = Q @ W^T ==============
// grid 128, block 256 (8 waves), 64 rows per block.
__global__ void __launch_bounds__(256) opq_zq_kernel(const float* __restrict__ codebooks,
                                                     const float* __restrict__ zrot,
                                                     const __bf16* __restrict__ wbf,
                                                     const int* __restrict__ idxbuf,
                                                     float* __restrict__ zq,
                                                     float* __restrict__ accum) {
    __shared__ __bf16 Qt[64 * 512];               // 64 KB gathered quantized tile
    int tid = threadIdx.x, lane = tid & 31, wid = tid >> 5;
    int row0 = blockIdx.x * 64;

    {   // phase 1: gather codebook rows (quantized) + commitment partial
        int lr = tid >> 2;
        int row = row0 + lr;
        float csum = 0.f;
#pragma unroll
        for (int mm = 0; mm < 2; ++mm) {
            int m = (tid & 3) * 2 + mm;
            int id = idxbuf[(size_t)row * Mm + m];
            const float* cbp = codebooks + ((size_t)m * Kk + id) * DSs;
            const float* zr  = zrot + (size_t)row * Dd + m * DSs;
            __bf16* qd = Qt + (size_t)lr * Dd + m * DSs;
#pragma unroll 4
            for (int d = 0; d < DSs; ++d) {
                float cv = cbp[d];
                float dz = zr[d] - cv;
                csum += dz * dz;
                qd[d] = f32_to_bf16(cv);
            }
        }
        for (int off = 16; off; off >>= 1) csum += __shfl_xor(csum, off, 32);
        if (lane == 0) atomicAdd(&accum[0], csum);
    }
    __syncthreads();

    // phase 2: z_q tile = Qt[64x512] @ W^T ; B column j = row j of W (contiguous in wbf)
    int wr = wid >> 1, wh = wid & 1;              // 16-row strip, 256-col half
    v8f zero = {0, 0, 0, 0, 0, 0, 0, 0};
    v8f acc[16];
#pragma unroll
    for (int i = 0; i < 16; ++i) acc[i] = zero;

    for (int k0 = 0; k0 < Dd; k0 += 32) {
        v16bf a = load_a_frag(Qt + (size_t)wr * 16 * Dd + k0, Dd, lane);
#pragma unroll
        for (int nt = 0; nt < 16; ++nt) {
            int col = wh * 256 + nt * 16;
            v16bf b = load_b_frag(wbf + (size_t)col * Dd + k0, Dd, lane);
            acc[nt] = wmma_bf16(a, b, acc[nt]);
        }
    }
    int rh = (lane >> 4) ? 8 : 0;
#pragma unroll
    for (int nt = 0; nt < 16; ++nt) {
#pragma unroll
        for (int r = 0; r < 8; ++r) {
            int row = row0 + wr * 16 + r + rh;
            int col = wh * 256 + nt * 16 + (lane & 15);
            zq[(size_t)row * Dd + col] = acc[nt][r];
        }
    }
}

// ============== kernel 4: ortho loss ||W^T W - I||_F^2 ==============
__global__ void __launch_bounds__(256) opq_ortho_kernel(const float* __restrict__ W,
                                                        float* __restrict__ accum) {
    int p = blockIdx.x * 256 + threadIdx.x;       // 262144 (i,j) pairs
    int i = p >> 9, j = p & 511;
    float dot = 0.f;
#pragma unroll 4
    for (int d = 0; d < Dd; ++d) dot += W[(size_t)d * Dd + i] * W[(size_t)d * Dd + j];
    float diff = dot - ((i == j) ? 1.f : 0.f);
    float v = diff * diff;
    int lane = threadIdx.x & 31;
    for (int off = 16; off; off >>= 1) v += __shfl_xor(v, off, 32);
    if (lane == 0) atomicAdd(&accum[1], v);
}

// ============== kernel 5: finalize scalars (perplexity, losses) ==============
__global__ void __launch_bounds__(256) opq_finalize_kernel(const int* __restrict__ counts,
                                                           const float* __restrict__ accum,
                                                           float* __restrict__ out_sc) {
    __shared__ float sm[8];
    int tid = threadIdx.x, lane = tid & 31, m = tid >> 5;
    float h = 0.f;
    for (int k = lane; k < Kk; k += 32) {
        float pr = (float)counts[m * Kk + k] * (1.0f / (float)Nn);
        h -= pr * logf(pr + 1e-10f);
    }
    for (int off = 16; off; off >>= 1) h += __shfl_xor(h, off, 32);
    if (lane == 0) sm[m] = expf(h);
    __syncthreads();
    if (tid == 0) {
        float pp = 0.f;
        for (int mm = 0; mm < 8; ++mm) pp += sm[mm];
        out_sc[0] = accum[0] * (1.0f / (float)(Nn * Dd));   // commitment mean over B*T*M*DS
        out_sc[1] = accum[1];                               // ortho loss
        out_sc[2] = pp * (1.0f / 8.0f);                     // perplexity
    }
}

// ---------------- host launcher ----------------
extern "C" void kernel_launch(void* const* d_in, const int* in_sizes, int n_in,
                              void* d_out, int out_size, void* d_ws, size_t ws_size,
                              hipStream_t stream) {
    const float* z         = (const float*)d_in[0];
    const float* W         = (const float*)d_in[1];
    const float* codebooks = (const float*)d_in[2];
    const float* gumbel    = (const float*)d_in[3];

    char* ws = (char*)d_ws;
    float*  zrot   = (float*)(ws + ZROT_OFF);
    __bf16* zn     = (__bf16*)(ws + ZN_OFF);
    __bf16* cnbf   = (__bf16*)(ws + CNBF_OFF);
    __bf16* wbf    = (__bf16*)(ws + WBF_OFF);
    int*    idxbuf = (int*)(ws + IDX_OFF);
    int*    counts = (int*)(ws + CNT_OFF);
    float*  accum  = (float*)(ws + ACC_OFF);

    float* out     = (float*)d_out;
    float* zq      = out;                                   // [N*D]
    float* out_idx = out + (size_t)Nn * Dd;                 // [N*M] as float
    float* out_sc  = out_idx + (size_t)Nn * Mm;             // 3 scalars

    hipMemsetAsync(ws + CNT_OFF, 0, (size_t)Mm * Kk * 4 + 16, stream);

    opq_prep_kernel<<<dim3(16 + 1024), dim3(256), 0, stream>>>(W, codebooks, cnbf, wbf);
    opq_rotate_kernel<<<dim3(Nn / 64, Dd / 64), dim3(256), 0, stream>>>(z, W, zrot, zn);
    opq_argmax_kernel<<<dim3((Nn / 16) * Mm / 4), dim3(128), 0, stream>>>(zn, cnbf, gumbel,
                                                                          idxbuf, out_idx, counts);
    opq_zq_kernel<<<dim3(Nn / 64), dim3(256), 0, stream>>>(codebooks, zrot, wbf, idxbuf, zq, accum);
    opq_ortho_kernel<<<dim3(1024), dim3(256), 0, stream>>>(W, accum);
    opq_finalize_kernel<<<dim3(1), dim3(256), 0, stream>>>(counts, accum, out_sc);
}
